// scModel_1211180777744
// MI455X (gfx1250) — matrix operand order repaired and checked by reading, not compiled, not run on hardware
//
#include <hip/hip_runtime.h>

// ---------------- model constants ----------------
#define BATCH   2
#define ENCL    1024
#define DECLEN  10000
#define DMODEL  512
#define NHEAD   8
#define HDIM    64
#define FFDIM   2048
#define NBIN    100
#define MFEAT   266
#define PADID   10001
#define MASKID  10002
#define VROW    (ENCL + DECLEN)
#define EPSK    1e-4f

// ---------------- WMMA types ----------------
typedef __attribute__((ext_vector_type(16))) __bf16 v16bf;
typedef __attribute__((ext_vector_type(8)))  float  v8f;

// pack two f32 -> packed bf16 pair (round-half-up) with a single v_perm_b32
__device__ __forceinline__ unsigned pk2(float lo, float hi) {
  unsigned ulo = __builtin_bit_cast(unsigned, lo) + 0x8000u;
  unsigned uhi = __builtin_bit_cast(unsigned, hi) + 0x8000u;
  // dest bytes: {lo[2], lo[3], hi[2], hi[3]}  (S1 supplies sel 0-3, S0 sel 4-7)
  return __builtin_amdgcn_perm(uhi, ulo, 0x07060302u);
}

// ---------------- generic batched WMMA GEMM ----------------
// C[z] = act(alpha * op(A[z]) @ op(B[z]) + bias)
// z decomposed as (b = z / Hdiv, h = z % Hdiv); float-element strides.
struct GemmP {
  const float* A; const float* B; const float* bias; float* C;
  int M, N, K;
  int lda, ldb, ldc;
  long long sAb, sAh, sBb, sBh, sCb, sCh;
  int Hdiv;
  int transA, transB, act;    // act: 0 none, 1 relu, 2 gelu(tanh)
  int aVec4, bVec4;           // operand eligible for float4 fast path
  float alpha;
};

#define BM 128
#define BN 128
#define BK 32
#define APITCH (BK + 8)   // 40 bf16 = 80B  (16B-aligned rows)
#define BPITCH (BN + 8)   // 136 bf16 = 272B (16B-aligned rows)

__global__ __launch_bounds__(256) void gemm_bf16_kernel(GemmP p) {
  __shared__ __bf16 As[BM][APITCH];
  __shared__ __bf16 Bs[BK][BPITCH];

  int z  = blockIdx.z;
  int bz = z / p.Hdiv, hz = z % p.Hdiv;
  const float* A  = p.A + bz * p.sAb + hz * p.sAh;
  const float* Bm = p.B + bz * p.sBb + hz * p.sBh;
  float* C = p.C + bz * p.sCb + hz * p.sCh;

  int m0 = blockIdx.y * BM;
  int n0 = blockIdx.x * BN;
  int tid = threadIdx.x;
  int wave = tid >> 5, lane = tid & 31;

  v8f acc[8] = {};

  // per-thread staging coords
  int arow = tid >> 1;              // 0..127
  int acb  = (tid & 1) * 16;        // 0 or 16
  int brow = tid >> 3;              // 0..31
  int bcb  = (tid & 7) * 16;        // 0..112

  float ra[16], rb[16];             // software-pipeline registers

  auto loadA = [&](int k0) {
    int gm = m0 + arow;
    if (!p.transA && p.aVec4 && gm < p.M && (k0 + acb + 15) < p.K) {
      const float* src = A + (long long)gm * p.lda + k0 + acb;
      *(float4*)&ra[0]  = *(const float4*)(src + 0);
      *(float4*)&ra[4]  = *(const float4*)(src + 4);
      *(float4*)&ra[8]  = *(const float4*)(src + 8);
      *(float4*)&ra[12] = *(const float4*)(src + 12);
      if (k0 + 2 * BK <= p.K) __builtin_prefetch(src + BK, 0, 0);
    } else {
      #pragma unroll
      for (int i = 0; i < 16; ++i) {
        int gk = k0 + acb + i;
        float v = 0.f;
        if (gm < p.M && gk < p.K)
          v = p.transA ? A[(long long)gk * p.lda + gm]
                       : A[(long long)gm * p.lda + gk];
        ra[i] = v;
      }
    }
  };
  auto loadB = [&](int k0) {
    int gk = k0 + brow;
    if (!p.transB && p.bVec4 && gk < p.K && (n0 + bcb + 15) < p.N) {
      const float* src = Bm + (long long)gk * p.ldb + n0 + bcb;
      *(float4*)&rb[0]  = *(const float4*)(src + 0);
      *(float4*)&rb[4]  = *(const float4*)(src + 4);
      *(float4*)&rb[8]  = *(const float4*)(src + 8);
      *(float4*)&rb[12] = *(const float4*)(src + 12);
      if (k0 + 2 * BK <= p.K)
        __builtin_prefetch(src + (long long)BK * p.ldb, 0, 0);
    } else {
      #pragma unroll
      for (int i = 0; i < 16; ++i) {
        int gn = n0 + bcb + i;
        float v = 0.f;
        if (gk < p.K && gn < p.N)
          v = p.transB ? Bm[(long long)gn * p.ldb + gk]
                       : Bm[(long long)gk * p.ldb + gn];
        rb[i] = v;
      }
    }
  };

  loadA(0);
  loadB(0);

  for (int k0 = 0; k0 < p.K; k0 += BK) {
    // ---- pack regs -> LDS (always 2x b128 stores each) ----
    {
      uint4 q0, q1;
      q0.x = pk2(ra[0], ra[1]);   q0.y = pk2(ra[2], ra[3]);
      q0.z = pk2(ra[4], ra[5]);   q0.w = pk2(ra[6], ra[7]);
      q1.x = pk2(ra[8], ra[9]);   q1.y = pk2(ra[10], ra[11]);
      q1.z = pk2(ra[12], ra[13]); q1.w = pk2(ra[14], ra[15]);
      *(uint4*)&As[arow][acb + 0] = q0;
      *(uint4*)&As[arow][acb + 8] = q1;
      uint4 r0, r1;
      r0.x = pk2(rb[0], rb[1]);   r0.y = pk2(rb[2], rb[3]);
      r0.z = pk2(rb[4], rb[5]);   r0.w = pk2(rb[6], rb[7]);
      r1.x = pk2(rb[8], rb[9]);   r1.y = pk2(rb[10], rb[11]);
      r1.z = pk2(rb[12], rb[13]); r1.w = pk2(rb[14], rb[15]);
      *(uint4*)&Bs[brow][bcb + 0] = r0;
      *(uint4*)&Bs[brow][bcb + 8] = r1;
    }
    __syncthreads();

    // ---- issue next tile's global loads (overlap with WMMA below) ----
    if (k0 + BK < p.K) { loadA(k0 + BK); loadB(k0 + BK); }

    // ---- A fragment (CDNA5 layout): lane r holds M=wave*16+r ----
    {
      int r = lane & 15, half = lane >> 4;
      v16bf af;
      #pragma unroll
      for (int e = 0; e < 16; ++e) {
        int kk = (e < 8) ? (half * 8 + e) : (16 + half * 8 + (e - 8));
        af[e] = As[wave * 16 + r][kk];
      }
      #pragma unroll
      for (int j = 0; j < 8; ++j) {
        v16bf bfr;                     // B layout: lane = N, element = K (same order as A)
        #pragma unroll
        for (int e = 0; e < 16; ++e)
          bfr[e] = Bs[(e < 8) ? (half * 8 + e) : (16 + half * 8 + (e - 8))][j * 16 + r];
        acc[j] = __builtin_amdgcn_wmma_f32_16x16x32_bf16(
            false, af, false, bfr, (short)0, acc[j], false, false);
      }
    }
    __syncthreads();
  }

  // ---- store C: VGPR i -> M = i + 8*half ; N = lane&15 ----
  int r = lane & 15, half = lane >> 4;
  #pragma unroll
  for (int j = 0; j < 8; ++j) {
    int gn = n0 + j * 16 + r;
    if (gn >= p.N) continue;
    float bv = p.bias ? p.bias[gn] : 0.f;
    #pragma unroll
    for (int i = 0; i < 8; ++i) {
      int gm = m0 + wave * 16 + half * 8 + i;
      if (gm >= p.M) continue;
      float v = p.alpha * acc[j][i] + bv;
      if (p.act == 1) v = fmaxf(v, 0.f);
      else if (p.act == 2) {
        float x = v;
        v = 0.5f * x * (1.f + tanhf(0.7978845608f * (x + 0.044715f * x * x * x)));
      }
      C[(long long)gm * p.ldc + gn] = v;
    }
  }
}

// ---------------- token embedding (AutoDiscretization soft-bin) ----------------
__global__ void token_embed_kernel(const float* values, int valOffset, int Ltok,
                                   const float* w1, const float* b1,
                                   const float* w2, const float* b2,
                                   const float* emb, const float* mask_emb,
                                   const float* pad_emb,
                                   const float* pos_emb,   // nullable
                                   const int* gene_ids, const int* indexv,
                                   float* out) {
  int token = blockIdx.x;
  int b = token / Ltok, l = token % Ltok;
  float x = values[(long long)b * VROW + valOffset + l];

  __shared__ float h[NBIN], w[NBIN];
  __shared__ float mx, sm;
  int tid = threadIdx.x;

  for (int j = tid; j < NBIN; j += blockDim.x) {
    float t = x * w1[j] + b1[j];
    h[j] = (t >= 0.f) ? t : 0.1f * t;           // leaky_relu 0.1
  }
  __syncthreads();
  for (int j = tid; j < NBIN; j += blockDim.x) {
    float s = b2[j] + h[j];                     // BIN_ALPHA = 1.0
    for (int i = 0; i < NBIN; ++i) s += h[i] * w2[i * NBIN + j];
    w[j] = s;
  }
  __syncthreads();
  if (tid == 0) { float m = -1e30f; for (int j = 0; j < NBIN; ++j) m = fmaxf(m, w[j]); mx = m; }
  __syncthreads();
  for (int j = tid; j < NBIN; j += blockDim.x) w[j] = expf(w[j] - mx);
  __syncthreads();
  if (tid == 0) { float s = 0.f; for (int j = 0; j < NBIN; ++j) s += w[j]; sm = s; }
  __syncthreads();

  bool is_mask = (x == (float)MASKID);
  bool is_pad  = (x == (float)PADID);
  int g = 0;
  if (pos_emb) g = gene_ids[indexv[b] * ENCL + l];
  float inv = 1.f / sm;
  for (int d = tid; d < DMODEL; d += blockDim.x) {
    float e = 0.f;
    for (int i = 0; i < NBIN; ++i) e += w[i] * emb[i * DMODEL + d];
    e *= inv;
    if (is_mask) e = mask_emb[d];
    if (is_pad)  e = pad_emb[d];
    if (pos_emb) e += pos_emb[(long long)g * DMODEL + d];
    out[(long long)token * DMODEL + d] = e;
  }
}

// ---------------- scatter encoder outputs into decoder seq ----------------
__global__ void scatter_kernel(const float* encX, const int* gene_ids,
                               const int* indexv, float* dec) {
  int l = blockIdx.x, b = blockIdx.y;
  int g = gene_ids[indexv[b] * ENCL + l];
  if (g == PADID) return;
  const float* src = encX + ((long long)b * ENCL + l) * DMODEL;
  float* dst = dec + ((long long)b * DECLEN + g) * DMODEL;
  for (int d = threadIdx.x; d < DMODEL; d += blockDim.x) dst[d] = src[d];
}

// ---------------- layernorm (optional residual), in-place safe per-row ----------------
__global__ void ln_kernel(const float* a, const float* res,
                          const float* s, const float* bb, float* out) {
  long long row = blockIdx.x;
  const float* ap = a + row * DMODEL;
  const float* rp = res ? res + row * DMODEL : nullptr;
  __shared__ float red[256];
  int tid = threadIdx.x;
  float sum = 0.f;
  for (int d = tid; d < DMODEL; d += 256) sum += ap[d] + (rp ? rp[d] : 0.f);
  red[tid] = sum; __syncthreads();
  for (int o = 128; o > 0; o >>= 1) { if (tid < o) red[tid] += red[tid + o]; __syncthreads(); }
  float mean = red[0] / DMODEL; __syncthreads();
  float vs = 0.f;
  for (int d = tid; d < DMODEL; d += 256) {
    float v = ap[d] + (rp ? rp[d] : 0.f) - mean; vs += v * v;
  }
  red[tid] = vs; __syncthreads();
  for (int o = 128; o > 0; o >>= 1) { if (tid < o) red[tid] += red[tid + o]; __syncthreads(); }
  float rstd = rsqrtf(red[0] / DMODEL + 1e-5f); __syncthreads();
  for (int d = tid; d < DMODEL; d += 256) {
    float v = ap[d] + (rp ? rp[d] : 0.f);
    out[row * DMODEL + d] = (v - mean) * rstd * s[d] + bb[d];
  }
}

// ---------------- row softmax with scale (encoder attention) ----------------
__global__ void softmax_kernel(float* sc, int ncols, float scale) {
  long long row = blockIdx.x;
  float* p = sc + row * ncols;
  __shared__ float red[256];
  int tid = threadIdx.x;
  float m = -1e30f;
  for (int j = tid; j < ncols; j += 256) m = fmaxf(m, p[j] * scale);
  red[tid] = m; __syncthreads();
  for (int o = 128; o > 0; o >>= 1) { if (tid < o) red[tid] = fmaxf(red[tid], red[tid + o]); __syncthreads(); }
  m = red[0]; __syncthreads();
  float s = 0.f;
  for (int j = tid; j < ncols; j += 256) { float e = expf(p[j] * scale - m); p[j] = e; s += e; }
  red[tid] = s; __syncthreads();
  for (int o = 128; o > 0; o >>= 1) { if (tid < o) red[tid] += red[tid + o]; __syncthreads(); }
  float inv = 1.f / red[0]; __syncthreads();
  for (int j = tid; j < ncols; j += 256) p[j] *= inv;
}

// ---------------- FAVOR+ helpers ----------------
__global__ void sumsq_kernel(const float* x, float* di, int L, float coef, long long total) {
  long long i = (long long)blockIdx.x * blockDim.x + threadIdx.x;
  if (i >= total) return;
  long long b = i / ((long long)NHEAD * L);
  long long rem = i % ((long long)NHEAD * L);
  long long h = rem / L, l = rem % L;
  const float* p = x + ((long long)b * L + l) * DMODEL + h * HDIM;
  float s = 0.f;
  for (int d = 0; d < HDIM; ++d) s += p[d] * p[d];
  di[i] = coef * s;
}

__global__ void qexp_kernel(float* qd, const float* qdi, float ratio) {
  long long row = blockIdx.x;
  float* p = qd + row * MFEAT;
  __shared__ float red[128];
  int tid = threadIdx.x;
  float m = -1e30f;
  for (int j = tid; j < MFEAT; j += 128) m = fmaxf(m, p[j]);
  red[tid] = m; __syncthreads();
  for (int o = 64; o > 0; o >>= 1) { if (tid < o) red[tid] = fmaxf(red[tid], red[tid + o]); __syncthreads(); }
  m = red[0];
  float di = qdi[row];
  __syncthreads();
  for (int j = tid; j < MFEAT; j += 128) p[j] = ratio * (expf(p[j] - di - m) + EPSK);
}

__global__ void kmax_kernel(const float* kd, float* kmax, int L) {
  int bh = blockIdx.x;
  const float* p = kd + (long long)bh * L * MFEAT;
  __shared__ float red[256];
  int tid = threadIdx.x;
  float m = -1e30f;
  long long n = (long long)L * MFEAT;
  for (long long j = tid; j < n; j += 256) m = fmaxf(m, p[j]);
  red[tid] = m; __syncthreads();
  for (int o = 128; o > 0; o >>= 1) { if (tid < o) red[tid] = fmaxf(red[tid], red[tid + o]); __syncthreads(); }
  if (tid == 0) kmax[bh] = red[0];
}

__global__ void kexp_kernel(float* kd, const float* kdi, const float* kmax, int L, float ratio) {
  long long row = blockIdx.x;
  int bh = (int)(row / L);
  float m = kmax[bh], di = kdi[row];
  float* p = kd + row * MFEAT;
  for (int j = threadIdx.x; j < MFEAT; j += 128) p[j] = ratio * (expf(p[j] - di - m) + EPSK);
}

__global__ void ksum_kernel(const float* kp, float* ksum, int L) {
  int bh = blockIdx.x;
  const float* p = kp + (long long)bh * L * MFEAT;
  for (int m = threadIdx.x; m < MFEAT; m += blockDim.x) {
    float s = 0.f;
    for (int l = 0; l < L; ++l) s += p[(long long)l * MFEAT + m];
    ksum[bh * MFEAT + m] = s;
  }
}

__global__ void dinv_scale_kernel(const float* qp, const float* ksum, float* o, int L) {
  long long row = blockIdx.x;               // b*H*L + h*L + l
  int bh = (int)(row / L);
  int l = (int)(row % L);
  int b = bh / NHEAD, h = bh % NHEAD;
  const float* qrow = qp + row * MFEAT;
  const float* ks = ksum + bh * MFEAT;
  __shared__ float red[64];
  int tid = threadIdx.x;
  float s = 0.f;
  for (int j = tid; j < MFEAT; j += 64) s += qrow[j] * ks[j];
  red[tid] = s; __syncthreads();
  for (int o2 = 32; o2 > 0; o2 >>= 1) { if (tid < o2) red[tid] += red[tid + o2]; __syncthreads(); }
  float dinv = 1.f / red[0]; __syncthreads();
  float* op = o + ((long long)b * L + l) * DMODEL + h * HDIM;
  if (tid < HDIM) op[tid] *= dinv;
}

__global__ void add_kernel(float* a, const float* b, long long n) {
  long long i = (long long)blockIdx.x * blockDim.x + threadIdx.x;
  if (i < n) a[i] += b[i];
}

// ---------------- host orchestration ----------------
extern "C" void kernel_launch(void* const* d_in, const int* in_sizes, int n_in,
                              void* d_out, int out_size, void* d_ws, size_t ws_size,
                              hipStream_t stream) {
  (void)in_sizes; (void)n_in; (void)out_size; (void)ws_size;
  auto F = [&](int i) { return (const float*)d_in[i]; };
  auto I = [&](int i) { return (const int*)d_in[i]; };

  const float* values = F(0);
  const float* tw1 = F(1); const float* tb1 = F(2);
  const float* tw2 = F(3); const float* tb2 = F(4);
  const float* temb = F(5); const float* tmask = F(6); const float* tpad = F(7);
  const float* pos_emb = F(8);
  const int encBase = 9;                 // 4 layers x 16 tensors
  const int perfBase = 9 + 4 * 16;       // 2 layers x 14 tensors
  const float* to_w = F(perfBase + 2 * 14);
  const float* to_b = F(perfBase + 2 * 14 + 1);
  const float* out_w = F(perfBase + 2 * 14 + 2);
  const float* out_b = F(perfBase + 2 * 14 + 3);
  const int* indexv = I(perfBase + 2 * 14 + 4);
  const int* gene_ids = I(perfBase + 2 * 14 + 5);

  // workspace carve (floats)
  float* wsp = (float*)d_ws;
  const long long ENCXY = (long long)BATCH * ENCL * DMODEL;
  const long long DECSZ = (long long)BATCH * DECLEN * DMODEL;
  const long long QPSZ  = (long long)BATCH * NHEAD * DECLEN * MFEAT;
  const long long FF1SZ = (long long)BATCH * DECLEN * FFDIM;
  float* encX = wsp;              wsp += ENCXY;
  float* decA = wsp;              wsp += DECSZ;   // decoder residual stream (pre to_out)
  float* decT = wsp;              wsp += DECSZ;   // tmp / LN output / attn-o
  float* dq   = wsp;              wsp += DECSZ;
  float* dk   = wsp;              wsp += DECSZ;
  float* dv   = wsp;              wsp += DECSZ;
  float* qpB  = wsp;              wsp += QPSZ;    // also encoder FF hidden
  float* kpB  = wsp;              wsp += QPSZ;
  float* ff1  = wsp;              wsp += FF1SZ;   // also encoder attn scores
  float* qdi  = wsp;              wsp += (long long)BATCH * NHEAD * DECLEN;
  float* kdi  = wsp;              wsp += (long long)BATCH * NHEAD * DECLEN;
  float* kmx  = wsp;              wsp += BATCH * NHEAD;
  float* ksm  = wsp;              wsp += BATCH * NHEAD * MFEAT;
  float* ctx  = wsp;              wsp += (long long)BATCH * NHEAD * MFEAT * HDIM;

  auto gemm = [&](const float* A, const float* Bm, const float* bias, float* C,
                  int M, int N, int K, int lda, int ldb, int ldc,
                  long long sAb, long long sAh, long long sBb, long long sBh,
                  long long sCb, long long sCh, int Z, int Hdiv,
                  int transA, int transB, int act, float alpha) {
    GemmP p;
    p.A = A; p.B = Bm; p.bias = bias; p.C = C;
    p.M = M; p.N = N; p.K = K; p.lda = lda; p.ldb = ldb; p.ldc = ldc;
    p.sAb = sAb; p.sAh = sAh; p.sBb = sBb; p.sBh = sBh; p.sCb = sCb; p.sCh = sCh;
    p.Hdiv = Hdiv; p.transA = transA; p.transB = transB; p.act = act;
    p.aVec4 = (!transA) && ((lda & 3) == 0) && ((((uintptr_t)A) & 15) == 0);
    p.bVec4 = (!transB) && ((ldb & 3) == 0) && ((((uintptr_t)Bm) & 15) == 0);
    p.alpha = alpha;
    dim3 g((N + BN - 1) / BN, (M + BM - 1) / BM, Z);
    gemm_bf16_kernel<<<g, dim3(256), 0, stream>>>(p);
  };

  // ================= encoder =================
  token_embed_kernel<<<BATCH * ENCL, 128, 0, stream>>>(
      values, 0, ENCL, tw1, tb1, tw2, tb2, temb, tmask, tpad,
      pos_emb, gene_ids, indexv, encX);

  const int ME = BATCH * ENCL;
  for (int layer = 0; layer < 4; ++layer) {
    const int b0 = encBase + layer * 16;
    const float *wq = F(b0+0), *bq = F(b0+1), *wk = F(b0+2), *bk = F(b0+3);
    const float *wv = F(b0+4), *bv = F(b0+5), *wo = F(b0+6), *bo = F(b0+7);
    const float *l1s = F(b0+8), *l1b = F(b0+9), *l2s = F(b0+10), *l2b = F(b0+11);
    const float *fw1 = F(b0+12), *fb1 = F(b0+13), *fw2 = F(b0+14), *fb2 = F(b0+15);

    gemm(encX, wq, bq, dq, ME, DMODEL, DMODEL, DMODEL, DMODEL, DMODEL,
         0,0,0,0,0,0, 1, 1, 0, 0, 0, 1.f);
    gemm(encX, wk, bk, dk, ME, DMODEL, DMODEL, DMODEL, DMODEL, DMODEL,
         0,0,0,0,0,0, 1, 1, 0, 0, 0, 1.f);
    gemm(encX, wv, bv, dv, ME, DMODEL, DMODEL, DMODEL, DMODEL, DMODEL,
         0,0,0,0,0,0, 1, 1, 0, 0, 0, 1.f);
    // scores[b,h] = q_head @ k_head^T  (M=N=1024, K=64)
    gemm(dq, dk, nullptr, ff1, ENCL, ENCL, HDIM, DMODEL, DMODEL, ENCL,
         (long long)ENCL*DMODEL, HDIM, (long long)ENCL*DMODEL, HDIM,
         (long long)NHEAD*ENCL*ENCL, (long long)ENCL*ENCL,
         BATCH*NHEAD, NHEAD, 0, 1, 0, 1.f);
    softmax_kernel<<<BATCH*NHEAD*ENCL, 256, 0, stream>>>(ff1, ENCL, 0.125f);
    // o[b,h] = att @ v_head -> decT (head-strided)
    gemm(ff1, dv, nullptr, decT, ENCL, HDIM, ENCL, ENCL, DMODEL, DMODEL,
         (long long)NHEAD*ENCL*ENCL, (long long)ENCL*ENCL,
         (long long)ENCL*DMODEL, HDIM, (long long)ENCL*DMODEL, HDIM,
         BATCH*NHEAD, NHEAD, 0, 0, 0, 1.f);
    gemm(decT, wo, bo, dq, ME, DMODEL, DMODEL, DMODEL, DMODEL, DMODEL,
         0,0,0,0,0,0, 1, 1, 0, 0, 0, 1.f);
    ln_kernel<<<ME, 256, 0, stream>>>(encX, dq, l1s, l1b, encX);
    gemm(encX, fw1, fb1, qpB, ME, FFDIM, DMODEL, DMODEL, FFDIM, FFDIM,
         0,0,0,0,0,0, 1, 1, 0, 0, 1, 1.f);               // relu
    gemm(qpB, fw2, fb2, dq, ME, DMODEL, FFDIM, FFDIM, DMODEL, DMODEL,
         0,0,0,0,0,0, 1, 1, 0, 0, 0, 1.f);
    ln_kernel<<<ME, 256, 0, stream>>>(encX, dq, l2s, l2b, encX);
  }

  // ================= decoder assembly =================
  token_embed_kernel<<<BATCH * DECLEN, 128, 0, stream>>>(
      values, ENCL, DECLEN, tw1, tb1, tw2, tb2, temb, tmask, tpad,
      nullptr, gene_ids, indexv, decA);
  scatter_kernel<<<dim3(ENCL, BATCH), 128, 0, stream>>>(encX, gene_ids, indexv, decA);

  const int MD = BATCH * DECLEN;
  // dec = dec @ to_out.w + b  -> decT becomes the residual stream; decA is scratch
  gemm(decA, to_w, to_b, decT, MD, DMODEL, DMODEL, DMODEL, DMODEL, DMODEL,
       0,0,0,0,0,0, 1, 1, 0, 0, 0, 1.f);
  float* cur = decT;
  float* tmpA = decA;

  const float dn = 0.3535533906f;                   // 64^-0.25
  const float ratio = 0.0613138f;                   // 266^-0.5
  const long long BHL = (long long)BATCH * NHEAD * DECLEN;

  // ================= performer layers =================
  for (int layer = 0; layer < 2; ++layer) {
    const int b0 = perfBase + layer * 14;
    const float *lnas = F(b0+0), *lnab = F(b0+1);
    const float *wq = F(b0+2), *wk = F(b0+3), *wv = F(b0+4);
    const float *wo = F(b0+5), *bo = F(b0+6), *proj = F(b0+7);
    const float *lnfs = F(b0+8), *lnfb = F(b0+9);
    const float *fw1 = F(b0+10), *fb1 = F(b0+11), *fw2 = F(b0+12), *fb2 = F(b0+13);

    ln_kernel<<<MD, 256, 0, stream>>>(cur, nullptr, lnas, lnab, tmpA);
    gemm(tmpA, wq, nullptr, dq, MD, DMODEL, DMODEL, DMODEL, DMODEL, DMODEL,
         0,0,0,0,0,0, 1, 1, 0, 0, 0, 1.f);
    gemm(tmpA, wk, nullptr, dk, MD, DMODEL, DMODEL, DMODEL, DMODEL, DMODEL,
         0,0,0,0,0,0, 1, 1, 0, 0, 0, 1.f);
    gemm(tmpA, wv, nullptr, dv, MD, DMODEL, DMODEL, DMODEL, DMODEL, DMODEL,
         0,0,0,0,0,0, 1, 1, 0, 0, 0, 1.f);
    // qd/kd = (x_head @ proj^T) * dn  (M=10000, N=266, K=64)
    gemm(dq, proj, nullptr, qpB, DECLEN, MFEAT, HDIM, DMODEL, HDIM, MFEAT,
         (long long)DECLEN*DMODEL, HDIM, 0, 0,
         (long long)NHEAD*DECLEN*MFEAT, (long long)DECLEN*MFEAT,
         BATCH*NHEAD, NHEAD, 0, 1, 0, dn);
    gemm(dk, proj, nullptr, kpB, DECLEN, MFEAT, HDIM, DMODEL, HDIM, MFEAT,
         (long long)DECLEN*DMODEL, HDIM, 0, 0,
         (long long)NHEAD*DECLEN*MFEAT, (long long)DECLEN*MFEAT,
         BATCH*NHEAD, NHEAD, 0, 1, 0, dn);
    sumsq_kernel<<<(int)((BHL + 255) / 256), 256, 0, stream>>>(dq, qdi, DECLEN, 0.5f*dn*dn, BHL);
    sumsq_kernel<<<(int)((BHL + 255) / 256), 256, 0, stream>>>(dk, kdi, DECLEN, 0.5f*dn*dn, BHL);
    qexp_kernel<<<(int)BHL, 128, 0, stream>>>(qpB, qdi, ratio);
    kmax_kernel<<<BATCH*NHEAD, 256, 0, stream>>>(kpB, kmx, DECLEN);
    kexp_kernel<<<(int)BHL, 128, 0, stream>>>(kpB, kdi, kmx, DECLEN, ratio);
    ksum_kernel<<<BATCH*NHEAD, 256, 0, stream>>>(kpB, ksm, DECLEN);
    // ctx[b,h] = kp^T @ v_head  (M=266, N=64, K=10000, transA)
    gemm(kpB, dv, nullptr, ctx, MFEAT, HDIM, DECLEN, MFEAT, DMODEL, HDIM,
         (long long)NHEAD*DECLEN*MFEAT, (long long)DECLEN*MFEAT,
         (long long)DECLEN*DMODEL, HDIM,
         (long long)NHEAD*MFEAT*HDIM, (long long)MFEAT*HDIM,
         BATCH*NHEAD, NHEAD, 1, 0, 0, 1.f);
    // o[b,h] = qp @ ctx -> tmpA (head-strided)
    gemm(qpB, ctx, nullptr, tmpA, DECLEN, HDIM, MFEAT, MFEAT, HDIM, DMODEL,
         (long long)NHEAD*DECLEN*MFEAT, (long long)DECLEN*MFEAT,
         (long long)NHEAD*MFEAT*HDIM, (long long)MFEAT*HDIM,
         (long long)DECLEN*DMODEL, HDIM,
         BATCH*NHEAD, NHEAD, 0, 0, 0, 1.f);
    dinv_scale_kernel<<<(int)BHL, 64, 0, stream>>>(qpB, ksm, tmpA, DECLEN);
    gemm(tmpA, wo, bo, dq, MD, DMODEL, DMODEL, DMODEL, DMODEL, DMODEL,
         0,0,0,0,0,0, 1, 1, 0, 0, 0, 1.f);
    add_kernel<<<(int)(((long long)MD*DMODEL + 255) / 256), 256, 0, stream>>>(cur, dq, (long long)MD*DMODEL);
    // FF block (pre-LN, gelu)
    ln_kernel<<<MD, 256, 0, stream>>>(cur, nullptr, lnfs, lnfb, tmpA);
    gemm(tmpA, fw1, fb1, ff1, MD, FFDIM, DMODEL, DMODEL, FFDIM, FFDIM,
         0,0,0,0,0,0, 1, 1, 0, 0, 2, 1.f);              // gelu
    gemm(ff1, fw2, fb2, dq, MD, DMODEL, FFDIM, FFDIM, DMODEL, DMODEL,
         0,0,0,0,0,0, 1, 1, 0, 0, 0, 1.f);
    add_kernel<<<(int)(((long long)MD*DMODEL + 255) / 256), 256, 0, stream>>>(cur, dq, (long long)MD*DMODEL);
  }

  // ================= output head =================
  gemm(cur, out_w, out_b, (float*)d_out, MD, 100, DMODEL, DMODEL, 100, 100,
       0,0,0,0,0,0, 1, 1, 0, 0, 0, 1.f);
}
